// AudioTransformer_73065983639710
// MI455X (gfx1250) — matrix operand order repaired and checked by reading, not compiled, not run on hardware
//
#include <hip/hip_runtime.h>
#include <math.h>

// ===========================================================================
// AudioTransformer on MI455X (gfx1250, wave32).
// All matmul-like ops go through v_wmma_f32_16x16x32_bf16 with b128
// fragment loads:
//   A 16x32 bf16 : m = lane%16, K(e) = e + (e&8) + 8*(lane>=16)
//                  -> two contiguous 8-half chunks at kk+8*half and +16
//   B 32x16 bf16 : n = lane%16, K(e) = e + (e&8) + 8*(lane>=16)
//                  -> two contiguous 8-half chunks at kk+8*half and +16,
//                     provided B is stored N-major (N x K "transposed")
//   C/D 16x16 f32: n = lane%16, m = i + 8*(lane>=16) for acc[i]
// Weights are pre-packed to (N,K) bf16; V is transposed+zero-padded so the
// attention PV matmul streams contiguous keys. The attention Q tile is
// staged into LDS via GLOBAL_LOAD_ASYNC_TO_LDS_B128 (ASYNCcnt) so phase-1
// A-fragments come from ds_load_b128. Needs ~70 MB workspace.
// ===========================================================================

typedef __bf16 bf16;
typedef __attribute__((ext_vector_type(16))) __bf16 v16bf;
typedef __attribute__((ext_vector_type(8)))  __bf16 v8bf;
typedef __attribute__((ext_vector_type(8)))  float  v8f;

static __device__ __forceinline__ v8bf ld8(const bf16* p) {
  return *(const v8bf*)p;                    // 16B -> global_load_b128 / ds_load_b128
}
static __device__ __forceinline__ v16bf combine(v8bf lo, v8bf hi) {
  return __builtin_shufflevector(lo, hi, 0, 1, 2, 3, 4, 5, 6, 7,
                                 8, 9, 10, 11, 12, 13, 14, 15);
}
static __device__ __forceinline__ v8f wmma_bf16(v16bf a, v16bf b, v8f c) {
  return __builtin_amdgcn_wmma_f32_16x16x32_bf16(false, a, false, b, (short)0, c,
                                                 false, false);
}

// ---------------------------------------------------------------------------
// WMMA GEMM: C[MxN] = relu?(A[MxK] * B + bias). A bf16 row-major (lda=K).
// Bt bf16 stored N-major: Bt[n*K + k]. C fp32 with row stride ldc; optional
// bf16 mirror Cb. Block = 8 waves as 2(M) x 4(N); wave tile = 16 x 64.
// Requires M % 32 == 0, K % 128 == 0, 16B-aligned buffers.
// ---------------------------------------------------------------------------
__global__ __launch_bounds__(256) void k_gemm_bf16(
    const bf16* __restrict__ A, const bf16* __restrict__ Bt,
    const float* __restrict__ bias, float* __restrict__ C, bf16* __restrict__ Cb,
    int M, int N, int K, int ldc, int relu)
{
  const int lane = threadIdx.x & 31;
  const int wave = threadIdx.x >> 5;
  const int half = lane >> 4;
  const int l16  = lane & 15;
  const int m0 = (blockIdx.y * 2 + (wave >> 2)) << 4;
  const int n0 = (blockIdx.x * 4 + (wave & 3)) << 6;
  if (m0 >= M || n0 >= N) return;            // wave-uniform

  const bf16* Ar = A + (size_t)(m0 + l16) * K + (half << 3);
  const bf16* Bp[4];
#pragma unroll
  for (int t = 0; t < 4; ++t) {
    int nl = n0 + (t << 4) + l16;
    if (nl > N - 1) nl = N - 1;              // clamp for ragged N (loads only)
    Bp[t] = Bt + (size_t)nl * K + (half << 3);
  }

  v8f acc[4];
#pragma unroll
  for (int t = 0; t < 4; ++t) acc[t] = (v8f){0.f,0.f,0.f,0.f,0.f,0.f,0.f,0.f};

#pragma unroll 4
  for (int kk = 0; kk < K; kk += 32) {
    __builtin_prefetch(Ar + kk + 256, 0, 3);               // near-cache prefetch
    const v16bf a = combine(ld8(Ar + kk), ld8(Ar + kk + 16));
#pragma unroll
    for (int t = 0; t < 4; ++t) {
      const v16bf b = combine(ld8(Bp[t] + kk), ld8(Bp[t] + kk + 16));
      acc[t] = wmma_bf16(a, b, acc[t]);
    }
  }

#pragma unroll
  for (int t = 0; t < 4; ++t) {
    const int ncol = n0 + (t << 4) + l16;
    if (ncol < N) {
      const float bs = bias ? bias[ncol] : 0.f;
#pragma unroll
      for (int i = 0; i < 8; ++i) {
        const int r = m0 + i + (half << 3);
        float v = acc[t][i] + bs;
        if (relu) v = v > 0.f ? v : 0.f;
        C[(size_t)r * ldc + ncol] = v;
        if (Cb) Cb[(size_t)r * ldc + ncol] = (bf16)v;
      }
    }
  }
}

// ---------------------------------------------------------------------------
// Local attention, LIMIT=128 (window 257). Block = (16-query tile, batch),
// 4 waves. Phase 0: async-copy Q tile to LDS. Phase 1: QK^T energies ->
// LDS fp32. Phase 2: banded softmax -> LDS bf16 probs. Phase 3: probs * V
// via Vt (B,E,S+256) zero-padded. OOB keys masked.
// ---------------------------------------------------------------------------
#define ATT_COLS 288   // 18 key tiles of 16; query row r valid cols [r, r+256]
#define ATT_PAD  128
#define ATT_E    512
__global__ __launch_bounds__(128) void k_attn(
    const bf16* __restrict__ Q, const bf16* __restrict__ Km,
    const bf16* __restrict__ Vt, float* __restrict__ O, int S, int E)
{
  __shared__ float sE[16][ATT_COLS + 16];
  __shared__ __align__(16) bf16 sP[16][ATT_COLS];
  __shared__ __align__(16) bf16 sQ[16][ATT_E];
  const int lane = threadIdx.x & 31;
  const int wave = threadIdx.x >> 5;
  const int half = lane >> 4;
  const int l16  = lane & 15;
  const int s0 = blockIdx.x << 4;
  const int b  = blockIdx.y;
  const size_t base = (size_t)b * S * E;
  const int SP = S + 2 * ATT_PAD;
  const float scale = rsqrtf((float)E);

  // ---- phase 0: async global->LDS copy of the 16 x E Q tile --------------
  {
    const int cpr = E >> 3;                  // 16B chunks per row
    const int nch = 16 * cpr;
    for (int id = threadIdx.x; id < nch; id += 128) {
      const int r  = id / cpr;
      const int c8 = (id % cpr) << 3;
      const bf16* g = Q + base + (size_t)(s0 + r) * E + c8;
      const unsigned ldsoff = (unsigned)(size_t)&sQ[r][c8];  // LDS = addr[31:0]
      asm volatile("global_load_async_to_lds_b128 %0, %1, off"
                   :: "v"(ldsoff), "v"((unsigned long long)(size_t)g)
                   : "memory");
    }
    asm volatile("s_wait_asynccnt 0" ::: "memory");
  }
  __syncthreads();

  // ---- phase 1: energies (Q from LDS, K rows from global) ----------------
  const bf16* Qr = &sQ[l16][0] + (half << 3);
  for (int j = wave; j < (ATT_COLS >> 4); j += 4) {
    int key = s0 - 128 + (j << 4) + l16;
    const int keyc = key < 0 ? 0 : (key >= S ? S - 1 : key);
    const bf16* Kr = Km + base + (size_t)keyc * E + (half << 3);
    v8f acc = {0.f, 0.f, 0.f, 0.f, 0.f, 0.f, 0.f, 0.f};
#pragma unroll 4
    for (int kk = 0; kk < E; kk += 32) {
      const v16bf a  = combine(ld8(Qr + kk), ld8(Qr + kk + 16));
      const v16bf bb = combine(ld8(Kr + kk), ld8(Kr + kk + 16));
      acc = wmma_bf16(a, bb, acc);
    }
#pragma unroll
    for (int i = 0; i < 8; ++i)
      sE[i + (half << 3)][(j << 4) + l16] = acc[i] * scale;
  }
  __syncthreads();

  // ---- phase 2: banded softmax, one thread per query row -----------------
  if (threadIdx.x < 16) {
    const int r = threadIdx.x;
    float mx = -3.0e38f;
    for (int c = r; c <= r + 256; ++c) {
      const int key = s0 - 128 + c;
      if (key >= 0 && key < S) { const float v = sE[r][c]; mx = v > mx ? v : mx; }
    }
    float sum = 0.f;
    for (int c = 0; c < ATT_COLS; ++c) {
      float v = 0.f;
      const int key = s0 - 128 + c;
      if (c >= r && c <= r + 256 && key >= 0 && key < S)
        v = __expf(sE[r][c] - mx);
      sE[r][c] = v;
      sum += v;
    }
    const float inv = 1.f / sum;
    for (int c = 0; c < ATT_COLS; ++c) sP[r][c] = (bf16)(sE[r][c] * inv);
  }
  __syncthreads();

  // ---- phase 3: O = probs * V (keys contiguous in padded Vt) -------------
  const bf16* Ap = &sP[l16][0] + (half << 3);      // ds_load_b128 chunks
  for (int nt = wave; nt < (E >> 4); nt += 4) {
    const int ncol = (nt << 4) + l16;
    // padded key index = (s0 - 128 + kc + kb) + 128 = s0 + kc + kb
    const bf16* Vr = Vt + ((size_t)b * E + ncol) * SP + s0 + (half << 3);
    v8f acc = {0.f, 0.f, 0.f, 0.f, 0.f, 0.f, 0.f, 0.f};
#pragma unroll 3
    for (int kc = 0; kc < ATT_COLS; kc += 32) {
      const v16bf a  = combine(ld8(Ap + kc), ld8(Ap + kc + 16));
      const v16bf bb = combine(ld8(Vr + kc), ld8(Vr + kc + 16));
      acc = wmma_bf16(a, bb, acc);
    }
#pragma unroll
    for (int i = 0; i < 8; ++i)
      O[base + (size_t)(s0 + i + (half << 3)) * E + ncol] = acc[i];
  }
}

// ---------------------------------------------------------------------------
// Layout / elementwise kernels
// ---------------------------------------------------------------------------
__global__ void k_transpose_ct(const float* __restrict__ x, float* __restrict__ y,
                               int Bn, int C, int T) {
  const long i = (long)blockIdx.x * blockDim.x + threadIdx.x;
  const long n = (long)Bn * C * T;
  if (i >= n) return;
  const int t = (int)(i % T);
  const long r = i / T;
  const int c = (int)(r % C);
  const int b = (int)(r / C);
  y[((long)b * T + t) * C + c] = x[i];
}

// W (K,N) f32 row-major -> Wt (N,K) bf16
__global__ void k_pack_wT(const float* __restrict__ w, bf16* __restrict__ wt,
                          int K, int N) {
  const long i = (long)blockIdx.x * blockDim.x + threadIdx.x;
  const long n = (long)N * K;
  if (i >= n) return;
  const int k = (int)(i % K);
  const int nn = (int)(i / K);
  wt[i] = (bf16)w[(long)k * N + nn];
}

// conv w (Cout,Cin,KS) f32 -> wt[co*(KS*Cin) + k*Cin + ci] bf16  (N-major)
__global__ void k_pack_conv_wT(const float* __restrict__ w, bf16* __restrict__ wt,
                               int Cout, int Cin, int KS) {
  const long i = (long)blockIdx.x * blockDim.x + threadIdx.x;
  const long n = (long)Cout * KS * Cin;
  if (i >= n) return;
  const int ci = (int)(i % Cin);
  const long r = i / Cin;
  const int k  = (int)(r % KS);
  const int co = (int)(r / KS);
  wt[i] = (bf16)w[((long)co * Cin + ci) * KS + k];
}

// dec w (Cin,Cout,4) f32 -> wt[co*(2*Cin) + rr]: rr<Cin tap0 else tap1 (N-major)
__global__ void k_pack_dec_wT(const float* __restrict__ w, bf16* __restrict__ wt,
                              int Cin, int Cout, int tap0, int tap1) {
  const long i = (long)blockIdx.x * blockDim.x + threadIdx.x;
  const long n = (long)Cout * 2 * Cin;
  if (i >= n) return;
  const int rr = (int)(i % (2 * Cin));
  const int co = (int)(i / (2 * Cin));
  const int ci = (rr < Cin) ? rr : rr - Cin;
  const int tap = (rr < Cin) ? tap0 : tap1;
  wt[i] = (bf16)w[((long)ci * Cout + co) * 4 + tap];
}

// v (B*S, E) bf16 -> vt (B, E, S+2P) bf16 zero-padded along keys
__global__ void k_vtrans_pad(const bf16* __restrict__ v, bf16* __restrict__ vt,
                             int Bn, int S, int E, int P) {
  const long i = (long)blockIdx.x * blockDim.x + threadIdx.x;
  const int SP = S + 2 * P;
  const long n = (long)Bn * E * SP;
  if (i >= n) return;
  const int sp = (int)(i % SP);
  const long r = i / SP;
  const int e = (int)(r % E);
  const int b = (int)(r / E);
  const int s = sp - P;
  vt[i] = (s >= 0 && s < S) ? v[((long)b * S + s) * E + e] : (bf16)0.f;
}

// act (B,L,Cin) f32 -> patches (B*L, 5*Cin) bf16 (k-major, zero-padded edges)
__global__ void k_im2col_enc(const float* __restrict__ a, bf16* __restrict__ p,
                             int Bn, int L, int Cin) {
  const long i = (long)blockIdx.x * blockDim.x + threadIdx.x;
  const long n = (long)Bn * L * 5 * Cin;
  if (i >= n) return;
  const int c = (int)(i % Cin);
  long r = i / Cin;
  const int k = (int)(r % 5);
  r /= 5;
  const int t = (int)(r % L);
  const int b = (int)(r / L);
  const int ts = t + k - 2;
  const float v = (ts >= 0 && ts < L) ? a[((long)b * L + ts) * Cin + c] : 0.f;
  p[i] = (bf16)v;
}

__global__ void k_maxpool2(const float* __restrict__ a, float* __restrict__ o,
                           int Bn, int Lout, int C) {
  const long i = (long)blockIdx.x * blockDim.x + threadIdx.x;
  const long n = (long)Bn * Lout * C;
  if (i >= n) return;
  const int c = (int)(i % C);
  const long r = i / C;
  const int t = (int)(r % Lout);
  const int b = (int)(r / Lout);
  const float* base = a + ((long)b * (2 * Lout) + 2 * t) * C + c;
  const float v0 = base[0], v1 = base[C];
  o[i] = v0 > v1 ? v0 : v1;
}

__global__ void k_add_pe(const float* __restrict__ h, float* __restrict__ hp,
                         bf16* __restrict__ hpb, int S, int E, long n) {
  const long i = (long)blockIdx.x * blockDim.x + threadIdx.x;
  if (i >= n) return;
  const int e = (int)(i % E);
  const long se = i / E;
  const int s = (int)(se % S);
  const int j2 = (e >> 1) << 1;
  const float div = __expf(-logf(10000.f) * (float)j2 / (float)E);
  const float ang = (float)s * div;
  const float pe = (e & 1) ? __cosf(ang) : __sinf(ang);
  const float v = h[i] + pe;
  hp[i] = v;
  hpb[i] = (bf16)v;
}

__global__ void k_add(const float* __restrict__ a, const float* __restrict__ b,
                      float* __restrict__ c, long n) {
  const long i = (long)blockIdx.x * blockDim.x + threadIdx.x;
  if (i < n) c[i] = a[i] + b[i];
}

// One row per block (E=512); writes fp32 y and bf16 yb
__global__ __launch_bounds__(256) void k_layernorm(
    const float* __restrict__ x, const float* __restrict__ g,
    const float* __restrict__ be, float* __restrict__ y, bf16* __restrict__ yb,
    int E)
{
  __shared__ float red[256];
  const int row = blockIdx.x;
  const int t = threadIdx.x;
  const float* xr = x + (size_t)row * E;
  float s = 0.f;
  for (int e = t; e < E; e += 256) s += xr[e];
  red[t] = s; __syncthreads();
  for (int o = 128; o > 0; o >>= 1) { if (t < o) red[t] += red[t + o]; __syncthreads(); }
  const float mean = red[0] / (float)E;
  __syncthreads();
  float s2 = 0.f;
  for (int e = t; e < E; e += 256) { const float d = xr[e] - mean; s2 += d * d; }
  red[t] = s2; __syncthreads();
  for (int o = 128; o > 0; o >>= 1) { if (t < o) red[t] += red[t + o]; __syncthreads(); }
  const float inv = rsqrtf(red[0] / (float)E + 1e-5f);
  float* yr = y + (size_t)row * E;
  bf16* ybr = yb + (size_t)row * E;
  for (int e = t; e < E; e += 256) {
    const float v = (xr[e] - mean) * inv * g[e] + be[e];
    yr[e] = v;
    ybr[e] = (bf16)v;
  }
}

// act (B,L,C) f32 -> patches (B*L, 2*C) bf16: halves from t+off0 / t+off1
__global__ void k_im2col_dec(const float* __restrict__ a, bf16* __restrict__ p,
                             int Bn, int L, int C, int off0, int off1) {
  const long i = (long)blockIdx.x * blockDim.x + threadIdx.x;
  const long n = (long)Bn * L * 2 * C;
  if (i >= n) return;
  const int c = (int)(i % C);
  long r = i / C;
  const int hh = (int)(r % 2);
  r /= 2;
  const int t = (int)(r % L);
  const int b = (int)(r / L);
  const int ts = t + (hh ? off1 : off0);
  const float v = (ts >= 0 && ts < L) ? a[((long)b * L + ts) * C + c] : 0.f;
  p[i] = (bf16)v;
}

// ===========================================================================
extern "C" void kernel_launch(void* const* d_in, const int* in_sizes, int n_in,
                              void* d_out, int out_size, void* d_ws, size_t ws_size,
                              hipStream_t stream)
{
  (void)in_sizes; (void)n_in; (void)out_size;
  const int Bn = 2, T = 4096, NM = 128, E = 512, S = 512, FFd = 2048;

  const float* x = (const float*)d_in[0];
  const float* ecw[3] = {(const float*)d_in[1], (const float*)d_in[2], (const float*)d_in[3]};
  const float* ecb[3] = {(const float*)d_in[4], (const float*)d_in[5], (const float*)d_in[6]};
  // layers: 7 + 14*l + {0:Wq 1:bq 2:Wk 3:bk 4:Wv 5:bv 6:W1 7:b1f 8:W2 9:b2f 10:g1 11:be1 12:g2 13:be2}
  const float* dcw[3] = {(const float*)d_in[63], (const float*)d_in[64], (const float*)d_in[65]};
  const float* dcb[3] = {(const float*)d_in[66], (const float*)d_in[67], (const float*)d_in[68]};
  float* out = (float*)d_out;

  // ---- workspace layout ---------------------------------------------------
  const size_t MB = (size_t)1 << 20;
  char* Wb = (char*)d_ws;
  size_t o = 0;
  auto AL = [&](size_t bytes) { size_t r = o; o += (bytes + 255) & ~(size_t)255; return r; };
  const size_t encW1 = AL((size_t)640 * 256 * 2);
  const size_t encW2 = AL((size_t)1280 * 512 * 2);
  const size_t encW3 = AL((size_t)2560 * 512 * 2);
  size_t wqo[4], wko[4], wvo[4], w1o[4], w2o[4];
  for (int l = 0; l < 4; ++l) {
    wqo[l] = AL((size_t)E * E * 2);
    wko[l] = AL((size_t)E * E * 2);
    wvo[l] = AL((size_t)E * E * 2);
    w1o[l] = AL((size_t)E * FFd * 2);
    w2o[l] = AL((size_t)FFd * E * 2);
  }
  const size_t d1e = AL((size_t)1024 * 512 * 2), d1o = AL((size_t)1024 * 512 * 2);
  const size_t d2e = AL((size_t)1024 * 512 * 2), d2o = AL((size_t)1024 * 512 * 2);
  const size_t d3e = AL((size_t)1024 * 1 * 2),   d3o = AL((size_t)1024 * 1 * 2);
  const size_t R0 = AL(12 * MB), R1 = AL(12 * MB), R2 = AL(12 * MB);
  const size_t HB = AL(2 * MB);
  if (o > ws_size) return;  // workspace too small: bail deterministically

  auto blocks = [](long n) { return dim3((unsigned)((n + 255) / 256)); };
  auto gemm = [&](const bf16* A, const bf16* Bt, const float* bias, float* C,
                  bf16* Cb, int M, int N, int K, int ldc, int relu) {
    dim3 g((unsigned)((N + 255) / 256), (unsigned)((M + 31) / 32));
    k_gemm_bf16<<<g, 256, 0, stream>>>(A, Bt, bias, C, Cb, M, N, K, ldc, relu);
  };
  auto F = [&](size_t off) { return (float*)(Wb + off); };
  auto H = [&](size_t off) { return (bf16*)(Wb + off); };

  // ---- pack weights to bf16, N-major (N,K) --------------------------------
  k_pack_conv_wT<<<blocks((long)256 * 5 * 128), 256, 0, stream>>>(ecw[0], H(encW1), 256, 128, 5);
  k_pack_conv_wT<<<blocks((long)512 * 5 * 256), 256, 0, stream>>>(ecw[1], H(encW2), 512, 256, 5);
  k_pack_conv_wT<<<blocks((long)512 * 5 * 512), 256, 0, stream>>>(ecw[2], H(encW3), 512, 512, 5);
  for (int l = 0; l < 4; ++l) {
    const int bi = 7 + 14 * l;
    k_pack_wT<<<blocks((long)E * E), 256, 0, stream>>>((const float*)d_in[bi + 0], H(wqo[l]), E, E);
    k_pack_wT<<<blocks((long)E * E), 256, 0, stream>>>((const float*)d_in[bi + 2], H(wko[l]), E, E);
    k_pack_wT<<<blocks((long)E * E), 256, 0, stream>>>((const float*)d_in[bi + 4], H(wvo[l]), E, E);
    k_pack_wT<<<blocks((long)E * FFd), 256, 0, stream>>>((const float*)d_in[bi + 6], H(w1o[l]), E, FFd);
    k_pack_wT<<<blocks((long)FFd * E), 256, 0, stream>>>((const float*)d_in[bi + 8], H(w2o[l]), FFd, E);
  }
  // transposed-conv taps: even out = taps (3,1) on x[i-1],x[i]; odd = (2,0) on x[i],x[i+1]
  k_pack_dec_wT<<<blocks((long)512 * 2 * 512), 256, 0, stream>>>(dcw[0], H(d1e), 512, 512, 3, 1);
  k_pack_dec_wT<<<blocks((long)512 * 2 * 512), 256, 0, stream>>>(dcw[0], H(d1o), 512, 512, 2, 0);
  k_pack_dec_wT<<<blocks((long)512 * 2 * 512), 256, 0, stream>>>(dcw[1], H(d2e), 512, 512, 3, 1);
  k_pack_dec_wT<<<blocks((long)512 * 2 * 512), 256, 0, stream>>>(dcw[1], H(d2o), 512, 512, 2, 0);
  k_pack_dec_wT<<<blocks((long)1 * 2 * 512), 256, 0, stream>>>(dcw[2], H(d3e), 512, 1, 3, 1);
  k_pack_dec_wT<<<blocks((long)1 * 2 * 512), 256, 0, stream>>>(dcw[2], H(d3o), 512, 1, 2, 0);

  // ---- encoder ------------------------------------------------------------
  k_transpose_ct<<<blocks((long)Bn * NM * T), 256, 0, stream>>>(x, F(R0), Bn, NM, T);
  k_im2col_enc<<<blocks((long)Bn * T * 5 * NM), 256, 0, stream>>>(F(R0), H(R1), Bn, T, NM);
  gemm(H(R1), H(encW1), ecb[0], F(R2), nullptr, Bn * T, 256, 5 * NM, 256, 1);
  k_maxpool2<<<blocks((long)Bn * 2048 * 256), 256, 0, stream>>>(F(R2), F(R0), Bn, 2048, 256);

  k_im2col_enc<<<blocks((long)Bn * 2048 * 5 * 256), 256, 0, stream>>>(F(R0), H(R1), Bn, 2048, 256);
  gemm(H(R1), H(encW2), ecb[1], F(R2), nullptr, Bn * 2048, 512, 5 * 256, 512, 1);
  k_maxpool2<<<blocks((long)Bn * 1024 * 512), 256, 0, stream>>>(F(R2), F(R0), Bn, 1024, 512);

  k_im2col_enc<<<blocks((long)Bn * 1024 * 5 * 512), 256, 0, stream>>>(F(R0), H(R1), Bn, 1024, 512);
  gemm(H(R1), H(encW3), ecb[2], F(R2), nullptr, Bn * 1024, 512, 5 * 512, 512, 1);
  k_maxpool2<<<blocks((long)Bn * 512 * 512), 256, 0, stream>>>(F(R2), F(HB), Bn, 512, 512);

  // ---- transformer --------------------------------------------------------
  float* hp_f = F(R0);              bf16* hp_b = H(R0 + 2 * MB);
  float* q_f  = F(R0 + 3 * MB);     float* kf  = F(R0 + 5 * MB);
  float* vf   = F(R0 + 7 * MB);     float* atf = F(R0 + 9 * MB);
  bf16* q_b = H(R1);                bf16* k_b = H(R1 + 1 * MB);
  bf16* v_b = H(R1 + 2 * MB);       float* r1 = F(R1 + 3 * MB);
  bf16* h_b = H(R1 + 5 * MB);       float* ff2f = F(R1 + 6 * MB);
  bf16* vt  = H(R1 + 8 * MB);       // (B,E,S+256) padded V^T, 1.5 MB
  float* ff1f = F(R2);              bf16* ff1b = H(R2 + 8 * MB);
  float* hF = F(HB);
  const long nSE = (long)Bn * S * E;
  const int Mrows = Bn * S;

  for (int l = 0; l < 4; ++l) {
    const int bi = 7 + 14 * l;
    const float* bq  = (const float*)d_in[bi + 1];
    const float* bk  = (const float*)d_in[bi + 3];
    const float* bv  = (const float*)d_in[bi + 5];
    const float* b1f = (const float*)d_in[bi + 7];
    const float* b2f = (const float*)d_in[bi + 9];
    const float* g1  = (const float*)d_in[bi + 10];
    const float* be1 = (const float*)d_in[bi + 11];
    const float* g2  = (const float*)d_in[bi + 12];
    const float* be2 = (const float*)d_in[bi + 13];

    k_add_pe<<<blocks(nSE), 256, 0, stream>>>(hF, hp_f, hp_b, S, E, nSE);
    gemm(hp_b, H(wqo[l]), bq, q_f, q_b, Mrows, E, E, E, 0);
    gemm(hp_b, H(wko[l]), bk, kf,  k_b, Mrows, E, E, E, 0);
    gemm(hp_b, H(wvo[l]), bv, vf,  v_b, Mrows, E, E, E, 0);
    k_vtrans_pad<<<blocks((long)Bn * E * (S + 256)), 256, 0, stream>>>(v_b, vt, Bn, S, E, 128);
    k_attn<<<dim3(S / 16, Bn), 128, 0, stream>>>(q_b, k_b, vt, atf, S, E);
    k_add<<<blocks(nSE), 256, 0, stream>>>(atf, hp_f, r1, nSE);
    k_layernorm<<<Mrows, 256, 0, stream>>>(r1, g1, be1, hF, h_b, E);
    gemm(h_b, H(w1o[l]), b1f, ff1f, ff1b, Mrows, FFd, E, FFd, 1);
    gemm(ff1b, H(w2o[l]), b2f, ff2f, nullptr, Mrows, E, FFd, E, 0);
    k_add<<<blocks(nSE), 256, 0, stream>>>(ff2f, hF, r1, nSE);
    k_layernorm<<<Mrows, 256, 0, stream>>>(r1, g2, be2, hF, h_b, E);
  }

  // ---- decoder (transposed convs as per-parity GEMMs) ---------------------
  // layer 1: (B,512,512) -> (B,1024,512), relu
  k_im2col_dec<<<blocks((long)Bn * 512 * 2 * 512), 256, 0, stream>>>(hF, H(R0), Bn, 512, 512, -1, 0);
  k_im2col_dec<<<blocks((long)Bn * 512 * 2 * 512), 256, 0, stream>>>(hF, H(R0 + 2 * MB), Bn, 512, 512, 0, 1);
  gemm(H(R0),          H(d1e), dcb[0], F(R1),       nullptr, Bn * 512, 512, 1024, 1024, 1);
  gemm(H(R0 + 2 * MB), H(d1o), dcb[0], F(R1) + 512, nullptr, Bn * 512, 512, 1024, 1024, 1);
  // layer 2: (B,1024,512) -> (B,2048,512), relu
  k_im2col_dec<<<blocks((long)Bn * 1024 * 2 * 512), 256, 0, stream>>>(F(R1), H(R0), Bn, 1024, 512, -1, 0);
  k_im2col_dec<<<blocks((long)Bn * 1024 * 2 * 512), 256, 0, stream>>>(F(R1), H(R0 + 4 * MB), Bn, 1024, 512, 0, 1);
  gemm(H(R0),          H(d2e), dcb[1], F(R2),       nullptr, Bn * 1024, 512, 1024, 1024, 1);
  gemm(H(R0 + 4 * MB), H(d2o), dcb[1], F(R2) + 512, nullptr, Bn * 1024, 512, 1024, 1024, 1);
  // layer 3: (B,2048,512) -> (B,4096,1), no relu, straight into d_out
  k_im2col_dec<<<blocks((long)Bn * 2048 * 2 * 512), 256, 0, stream>>>(F(R2), H(R0), Bn, 2048, 512, -1, 0);
  k_im2col_dec<<<blocks((long)Bn * 2048 * 2 * 512), 256, 0, stream>>>(F(R2), H(R1), Bn, 2048, 512, 0, 1);
  gemm(H(R0), H(d3e), dcb[2], out,     nullptr, Bn * 2048, 1, 1024, 2, 0);
  gemm(H(R1), H(d3o), dcb[2], out + 1, nullptr, Bn * 2048, 1, 1024, 2, 0);
}
